// GATLayer_3246995276176
// MI455X (gfx1250) — matrix operand order repaired and checked by reading, not compiled, not run on hardware
//
#include <hip/hip_runtime.h>

typedef __attribute__((ext_vector_type(2)))  float    v2f;
typedef __attribute__((ext_vector_type(4)))  float    v4f;
typedef __attribute__((ext_vector_type(8)))  float    v8f;
typedef __attribute__((ext_vector_type(16))) _Float16 v16h;
typedef __attribute__((ext_vector_type(4)))  int      v4i;

#define GN     8192
#define IN_C   256
#define OUT_C  64
#define ALPHA  0.2f
#define LOG2E  1.44269504088896340736f

// ---------------------------------------------------------------------------
// Kernel 1: h = x @ w via V_WMMA_F32_16X16X4_F32 (full fp32 path).
// Each wave owns one 16x16 output tile; K=256 -> 64 WMMA per wave.
// Emits h (f32, row-major, for the f_src/f_dst projections) and hT (f16,
// transposed 64x8192) so the attention GEMM's B operand is a contiguous
// 32B-per-lane load in contiguous k order (k = 16*hi + s), same as A.
// ---------------------------------------------------------------------------
__global__ __launch_bounds__(256) void k_gemm_h(const float* __restrict__ x,
                                                const float* __restrict__ w,
                                                float* __restrict__ h,
                                                _Float16* __restrict__ hT)
{
    const int lane   = threadIdx.x & 31;
    const int laneLo = lane & 15;
    const int hi     = lane >> 4;
    const int wave   = threadIdx.x >> 5;
    const int tile   = blockIdx.x * 8 + wave;      // 0..2047
    const int i0     = (tile >> 2) * 16;           // row tile
    const int c0     = (tile & 3) * 16;            // col tile

    v8f acc = {};
    // f32 A 16x4 layout: lanes 0-15 hold K=k,k+1 ; lanes 16-31 hold K=k+2,k+3
    const float* xr = x + (size_t)(i0 + laneLo) * IN_C + 2 * hi;
    // f32 B 4x16: element v, lane -> K = 2*hi + v, N = laneLo
    const float* wr = w + (size_t)(2 * hi) * OUT_C + c0 + laneLo;

#pragma unroll 4
    for (int k = 0; k < IN_C; k += 4) {
        v2f av = *(const v2f*)(xr + k);
        v2f bv;
        bv.x = wr[(size_t)k * OUT_C];
        bv.y = wr[(size_t)(k + 1) * OUT_C];
        acc = __builtin_amdgcn_wmma_f32_16x16x4_f32(false, av, false, bv,
                                                    (short)0, acc, false, false);
    }

    // C/D f32 16x16 layout: VGPR r -> M = r (lanes 0-15) / 8+r (lanes 16-31)
#pragma unroll
    for (int r = 0; r < 8; ++r) {
        const int row = i0 + hi * 8 + r;
        const int col = c0 + laneLo;
        const float v = acc[r];
        h[(size_t)row * OUT_C + col]  = v;
        hT[(size_t)col * GN + row]    = (_Float16)v;
    }
}

// ---------------------------------------------------------------------------
// Kernel 2: f_src[i] = h[i,:]·a_src ; f_dst[i] = h[i,:]·a_dst
// ---------------------------------------------------------------------------
__global__ __launch_bounds__(256) void k_proj(const float* __restrict__ h,
                                              const float* __restrict__ a,
                                              float* __restrict__ fsrc,
                                              float* __restrict__ fdst)
{
    const int i = blockIdx.x * 256 + threadIdx.x;
    const float* hr = h + (size_t)i * OUT_C;
    float s = 0.f, d = 0.f;
#pragma unroll 8
    for (int c = 0; c < OUT_C; ++c) {
        const float hv = hr[c];
        s = __builtin_fmaf(hv, a[c], s);
        d = __builtin_fmaf(hv, a[OUT_C + c], d);
    }
    fsrc[i] = s;
    fdst[i] = d;
}

// ---------------------------------------------------------------------------
// Kernel 3: gmax = max_j f_dst[j]  (one workgroup)
// ---------------------------------------------------------------------------
__global__ __launch_bounds__(256) void k_gmax(const float* __restrict__ fdst,
                                              float* __restrict__ gmax)
{
    __shared__ float red[256];
    float m = -3.0e38f;
    for (int j = threadIdx.x; j < GN; j += 256) m = fmaxf(m, fdst[j]);
    red[threadIdx.x] = m;
    __syncthreads();
    for (int s = 128; s > 0; s >>= 1) {
        if (threadIdx.x < s) red[threadIdx.x] = fmaxf(red[threadIdx.x], red[threadIdx.x + s]);
        __syncthreads();
    }
    if (threadIdx.x == 0) *gmax = red[0];
}

// ---------------------------------------------------------------------------
// Kernel 4: fused mask + leaky-relu + softmax + (attention @ h) + ELU.
// Block = one 16-row tile (8 waves x 1024-wide j-stripes). Per 32-j chunk:
// lane computes its 16 p-values in contiguous k order (k = 16*hi + s), same as B,
// then 4x v_wmma_f32_16x16x32_f16 against L2-resident hT tiles.
// Row stability via the gmax upper bound => single streaming pass over
// 'adjacent' (non-temporal: 256 MB read-once, don't pollute L2).
// ---------------------------------------------------------------------------
__global__ __launch_bounds__(256) void k_attn(const int* __restrict__ adj,
                                              const _Float16* __restrict__ hT,
                                              const float* __restrict__ fsrc,
                                              const float* __restrict__ fdst,
                                              const float* __restrict__ gmaxp,
                                              float* __restrict__ out)
{
    __shared__ float accLDS[16 * OUT_C];
    __shared__ float lLDS[16];

    const int lane   = threadIdx.x & 31;
    const int laneLo = lane & 15;
    const int hi     = lane >> 4;
    const int wave   = threadIdx.x >> 5;
    const int i0     = blockIdx.x * 16;

    for (int t = threadIdx.x; t < 16 * OUT_C; t += 256) accLDS[t] = 0.f;
    if (threadIdx.x < 16) lLDS[threadIdx.x] = 0.f;
    __syncthreads();

    const float s_my = fsrc[i0 + laneLo];
    const float gmax = *gmaxp;
    const float tg   = s_my + gmax;
    const float mL   = fmaxf(tg, ALPHA * tg) * LOG2E;   // row max upper bound (log2 units)

    v8f acc0 = {}, acc1 = {}, acc2 = {}, acc3 = {};
    float lsum = 0.f;

    const size_t adjRow = (size_t)(i0 + laneLo) * GN;
    const int    jw     = wave * 1024;

    for (int c = 0; c < 32; ++c) {
        const int jb = jw + c * 32;
        const int q0 = jb + hi * 16;   // A (contiguous k): lo lanes K 0-7 / hi lanes K 16-23
        const int q1 = q0 + 8;         //           lo lanes K 8-15 / hi lanes K 24-31

        const v4i A0 = __builtin_nontemporal_load((const v4i*)(adj + adjRow + q0));
        const v4i A1 = __builtin_nontemporal_load((const v4i*)(adj + adjRow + q0 + 4));
        const v4i A2 = __builtin_nontemporal_load((const v4i*)(adj + adjRow + q1));
        const v4i A3 = __builtin_nontemporal_load((const v4i*)(adj + adjRow + q1 + 4));

        const v4f D0 = *(const v4f*)(fdst + q0);
        const v4f D1 = *(const v4f*)(fdst + q0 + 4);
        const v4f D2 = *(const v4f*)(fdst + q1);
        const v4f D3 = *(const v4f*)(fdst + q1 + 4);

        v16h A;
#pragma unroll
        for (int e = 0; e < 4; ++e) {
            float t, ee, p;
            t  = s_my + D0[e];  ee = fmaxf(t, ALPHA * t);
            p  = (A0[e] > 0) ? __builtin_amdgcn_exp2f(__builtin_fmaf(ee, LOG2E, -mL)) : 0.f;
            lsum += p;  A[e] = (_Float16)p;

            t  = s_my + D1[e];  ee = fmaxf(t, ALPHA * t);
            p  = (A1[e] > 0) ? __builtin_amdgcn_exp2f(__builtin_fmaf(ee, LOG2E, -mL)) : 0.f;
            lsum += p;  A[4 + e] = (_Float16)p;

            t  = s_my + D2[e];  ee = fmaxf(t, ALPHA * t);
            p  = (A2[e] > 0) ? __builtin_amdgcn_exp2f(__builtin_fmaf(ee, LOG2E, -mL)) : 0.f;
            lsum += p;  A[8 + e] = (_Float16)p;

            t  = s_my + D3[e];  ee = fmaxf(t, ALPHA * t);
            p  = (A3[e] > 0) ? __builtin_amdgcn_exp2f(__builtin_fmaf(ee, LOG2E, -mL)) : 0.f;
            lsum += p;  A[12 + e] = (_Float16)p;
        }

        // B, contiguous k as in A: lane N = laneLo, lo lanes K=jb..jb+15, hi lanes +16
        const size_t hb = (size_t)jb + (size_t)hi * 16;
        const v16h B0 = *(const v16h*)(hT + (size_t)(laneLo +  0) * GN + hb);
        const v16h B1 = *(const v16h*)(hT + (size_t)(laneLo + 16) * GN + hb);
        const v16h B2 = *(const v16h*)(hT + (size_t)(laneLo + 32) * GN + hb);
        const v16h B3 = *(const v16h*)(hT + (size_t)(laneLo + 48) * GN + hb);

        acc0 = __builtin_amdgcn_wmma_f32_16x16x32_f16(false, A, false, B0, (short)0, acc0, false, false);
        acc1 = __builtin_amdgcn_wmma_f32_16x16x32_f16(false, A, false, B1, (short)0, acc1, false, false);
        acc2 = __builtin_amdgcn_wmma_f32_16x16x32_f16(false, A, false, B2, (short)0, acc2, false, false);
        acc3 = __builtin_amdgcn_wmma_f32_16x16x32_f16(false, A, false, B3, (short)0, acc3, false, false);
    }

    // merge the 8 wave partials through LDS float atomics (ds_add_f32)
#pragma unroll
    for (int r = 0; r < 8; ++r) {
        const int row = hi * 8 + r;
        atomicAdd(&accLDS[row * OUT_C +  0 + laneLo], acc0[r]);
        atomicAdd(&accLDS[row * OUT_C + 16 + laneLo], acc1[r]);
        atomicAdd(&accLDS[row * OUT_C + 32 + laneLo], acc2[r]);
        atomicAdd(&accLDS[row * OUT_C + 48 + laneLo], acc3[r]);
    }
    atomicAdd(&lLDS[laneLo], lsum);
    __syncthreads();

    // normalize + ELU + store
    for (int t = threadIdx.x; t < 16 * OUT_C; t += 256) {
        const int row = t >> 6;
        const float v = accLDS[t] / lLDS[row];
        const float r = (v > 0.f) ? v : (__builtin_amdgcn_exp2f(v * LOG2E) - 1.f);
        out[(size_t)i0 * OUT_C + t] = r;
    }
}

// ---------------------------------------------------------------------------
extern "C" void kernel_launch(void* const* d_in, const int* in_sizes, int n_in,
                              void* d_out, int out_size, void* d_ws, size_t ws_size,
                              hipStream_t stream)
{
    (void)in_sizes; (void)n_in; (void)out_size; (void)ws_size;

    const float* x   = (const float*)d_in[0];
    const int*   adj = (const int*)d_in[1];
    const float* w   = (const float*)d_in[2];
    const float* a   = (const float*)d_in[3];
    float*       out = (float*)d_out;

    char* ws = (char*)d_ws;
    float*    h    = (float*)ws;                                  // 2 MB
    _Float16* hT   = (_Float16*)(ws + (size_t)GN * OUT_C * 4);    // 1 MB
    float*    fsrc = (float*)(ws + (size_t)3 * 1024 * 1024);      // 32 KB
    float*    fdst = fsrc + GN;                                   // 32 KB
    float*    gmax = fdst + GN;                                   // 4 B

    k_gemm_h<<<256,    256, 0, stream>>>(x, w, h, hT);
    k_proj  <<<GN/256, 256, 0, stream>>>(h, a, fsrc, fdst);
    k_gmax  <<<1,      256, 0, stream>>>(fdst, gmax);
    k_attn  <<<GN/16,  256, 0, stream>>>(adj, hT, fsrc, fdst, gmax, out);
}
